// GBM_56719338111640
// MI455X (gfx1250) — matrix operand order repaired and checked by reading, not compiled, not run on hardware
//
#include <hip/hip_runtime.h>
#include <hip/hip_bf16.h>
#include <math.h>

typedef __attribute__((ext_vector_type(16))) __bf16 v16bf;
typedef __attribute__((ext_vector_type(8)))  __bf16 v8bf;
typedef __attribute__((ext_vector_type(4)))  __bf16 v4bf;
typedef __attribute__((ext_vector_type(8)))  float  v8f;

#define BATCH_   4
#define SEQ_     1024
#define R_TOT    4096      // BATCH*SEQ
#define LAT_     1024      // N_DIST*N_CAT
#define DM_      512
#define DI_      1024      // 2*DM
#define DST_     16
#define DTR_     32
#define NPROJ_   64        // DT_RANK + 2*D_STATE

union Frag16 { v16bf v; v8bf h[2]; };

__device__ __forceinline__ float silu_f(float x) { return x / (1.f + __expf(-x)); }

// ---------------------------------------------------------------------------
// bf16 WMMA GEMM: C[M,N] = A[M,K](LDA,bf16) * B[K,N](bf16) (+bias)(+C)(act)
// BM=128, BN=32*TN, BK=32. 256 threads = 8 waves (4 M-waves x 2 N-waves),
// wave tile 32 x (16*TN) = 2 x TN WMMA tiles.
// A tile staged global->LDS with GLOBAL_LOAD_ASYNC_TO_LDS_B128 (ASYNCcnt),
// B tile staged transposed ([n][k]) for contiguous fragment ds reads.
// N/K/LDA compile-time so the epilogue address math constant-folds.
// ACT: 0 = none, 1 = softplus.  ACCUM: C += A*B.
// ---------------------------------------------------------------------------
template<int TN, int N, int K, int LDA, int ACT, bool ACCUM, bool HAS_BIAS>
__global__ __launch_bounds__(256) void gemm_bf16(
    const __bf16* __restrict__ A, const __bf16* __restrict__ B,
    const float* __restrict__ bias, float* __restrict__ C)
{
  constexpr int BN = 32 * TN;
  __shared__ __align__(16) __bf16 sA[128][40]; // 80B rows (16B-aligned chunks)
  __shared__ __align__(16) __bf16 sB[BN][40];  // transposed [n][k]

  const int tid  = threadIdx.x;
  const int m0   = blockIdx.y * 128;
  const int n0   = blockIdx.x * BN;
  const int wave = tid >> 5, lane = tid & 31;
  const int wm   = wave >> 1, wn = wave & 1;
  const int r    = lane & 15, hi = lane >> 4;
  const unsigned long long abase = (unsigned long long)A;

  const v8f vzero = {0.f,0.f,0.f,0.f,0.f,0.f,0.f,0.f};
  v8f acc[2][TN];
#pragma unroll
  for (int ti = 0; ti < 2; ++ti)
#pragma unroll
    for (int tj = 0; tj < TN; ++tj) acc[ti][tj] = vzero;

#pragma unroll 1
  for (int k0 = 0; k0 < K; k0 += 32) {
    // ---- A tile 128x32 bf16: async copy, 16B per lane-op, 2 ops/thread ----
#pragma unroll
    for (int i = 0; i < 2; ++i) {
      int idx = tid + 256 * i;
      int row = idx >> 2, seg = (idx & 3) * 8;          // 8 bf16 = 16 bytes
      unsigned ldsaddr = (unsigned)(size_t)&sA[row][seg];
      int goff = (int)(((m0 + row) * LDA + k0 + seg) * (int)sizeof(__bf16));
      asm volatile("global_load_async_to_lds_b128 %0, %1, %2"
                   :: "v"(ldsaddr), "v"(goff), "s"(abase) : "memory");
    }
    // ---- B tile 32xBN bf16 -> LDS transposed ----
#pragma unroll
    for (int i = 0; i < BN / 64; ++i) {
      int idx = tid + 256 * i;
      int kk = idx / (BN / 8);
      int n8 = (idx % (BN / 8)) * 8;
      v8bf q = *(const v8bf*)(B + (size_t)(k0 + kk) * N + n0 + n8);
#pragma unroll
      for (int j = 0; j < 8; ++j) sB[n8 + j][kk] = q[j];
    }
    asm volatile("s_wait_asynccnt 0x0" ::: "memory");
    __syncthreads();

    // ---- fragments ----
    Frag16 a[2], b[TN];
#pragma unroll
    for (int t = 0; t < 2; ++t) {
      // A frag (16x32): lanes 0-15 K={0..7,16..23}, lanes 16-31 K={8..15,24..31}
      const __bf16* pa = &sA[wm * 32 + t * 16 + r][hi * 8];
      a[t].h[0] = *(const v8bf*)pa;
      a[t].h[1] = *(const v8bf*)(pa + 16);
    }
#pragma unroll
    for (int t = 0; t < TN; ++t) {
      // B frag (32x16): lanes 0-15 K={0..7,16..23}, lanes 16-31 K={8..15,24..31}, col = lane&15
      const __bf16* pb = &sB[wn * (TN * 16) + t * 16 + r][hi * 8];
      b[t].h[0] = *(const v8bf*)pb;
      b[t].h[1] = *(const v8bf*)(pb + 16);
    }
#pragma unroll
    for (int ti = 0; ti < 2; ++ti)
#pragma unroll
      for (int tj = 0; tj < TN; ++tj)
        acc[ti][tj] = __builtin_amdgcn_wmma_f32_16x16x32_bf16(
            false, a[ti].v, false, b[tj].v, (short)0, acc[ti][tj], false, false);
    __syncthreads();
  }

  // ---- epilogue: C layout — VGPR e: row = e + 8*hi, col = lane&15 ----
#pragma unroll
  for (int ti = 0; ti < 2; ++ti) {
#pragma unroll
    for (int tj = 0; tj < TN; ++tj) {
      int col = n0 + wn * (TN * 16) + tj * 16 + r;
      float bv = HAS_BIAS ? bias[col] : 0.f;
      float* cp = C + (size_t)(m0 + wm * 32 + ti * 16 + 8 * hi) * N + col;
#pragma unroll
      for (int e = 0; e < 8; ++e) {
        float v = acc[ti][tj][e] + bv;
        if (ACCUM) v += cp[(size_t)e * N];
        if (ACT == 1) v = (v > 20.f) ? v : log1pf(expf(v));
        cp[(size_t)e * N] = v;
      }
    }
  }
}

// ---------------------------------------------------------------------------
// f32 -> bf16 bulk convert (n multiple of 1024; grid = n/1024, block 256)
// ---------------------------------------------------------------------------
__global__ __launch_bounds__(256) void f2b_kernel(
    const float* __restrict__ src, __bf16* __restrict__ dst)
{
  size_t idx = (size_t)blockIdx.x * 256 + threadIdx.x;
  float4 f = *(const float4*)(src + idx * 4);
  v4bf p = { (__bf16)f.x, (__bf16)f.y, (__bf16)f.z, (__bf16)f.w };
  *(v4bf*)(dst + idx * 4) = p;
}

// ---------------------------------------------------------------------------
// RMSNorm: one wave per row of 512, 8 rows per block; bf16 output
// ---------------------------------------------------------------------------
__global__ __launch_bounds__(256) void rmsnorm_kernel(
    const float* __restrict__ h, const float* __restrict__ w,
    __bf16* __restrict__ hn)
{
  int row  = blockIdx.x * 8 + (threadIdx.x >> 5);
  int lane = threadIdx.x & 31;
  const float* p = h + (size_t)row * DM_ + lane * 16;
  float4 f[4];
#pragma unroll
  for (int i = 0; i < 4; ++i) f[i] = *(const float4*)(p + i * 4);
  float ss = 0.f;
#pragma unroll
  for (int i = 0; i < 4; ++i)
    ss += f[i].x * f[i].x + f[i].y * f[i].y + f[i].z * f[i].z + f[i].w * f[i].w;
#pragma unroll
  for (int off = 16; off > 0; off >>= 1) ss += __shfl_xor(ss, off, 32);
  float scale = rsqrtf(ss * (1.0f / DM_) + 1e-5f);
  __bf16* o = hn + (size_t)row * DM_ + lane * 16;
  const float* ww = w + lane * 16;
#pragma unroll
  for (int i = 0; i < 4; ++i) {
    float4 wv = *(const float4*)(ww + i * 4);
    v4bf r4 = { (__bf16)(f[i].x * scale * wv.x), (__bf16)(f[i].y * scale * wv.y),
                (__bf16)(f[i].z * scale * wv.z), (__bf16)(f[i].w * scale * wv.w) };
    *(v4bf*)(o + i * 4) = r4;
  }
}

// ---------------------------------------------------------------------------
// Causal depthwise conv (K=4) over seq + SiLU; dual f32 + bf16 output
// ---------------------------------------------------------------------------
__global__ __launch_bounds__(256) void conv_silu_kernel(
    const float* __restrict__ xz, const float* __restrict__ cw,
    const float* __restrict__ cb, float* __restrict__ xc,
    __bf16* __restrict__ xcb)
{
  int idx = blockIdx.x * 256 + threadIdx.x;   // over R_TOT * DI_
  int d   = idx & (DI_ - 1);
  int rs  = idx >> 10;                        // b*SEQ + s
  int s   = rs & (SEQ_ - 1);
  float4 w4 = *(const float4*)(cw + (size_t)d * 4);
  float acc = cb[d];
  const float* xrow = xz + (size_t)rs * (2 * DI_) + d;
  acc += xrow[0] * w4.w;
  if (s >= 1) acc += xrow[-(2 * DI_)]     * w4.z;
  if (s >= 2) acc += xrow[-(2 * DI_) * 2] * w4.y;
  if (s >= 3) acc += xrow[-(2 * DI_) * 3] * w4.x;
  float v = silu_f(acc);
  xc[idx]  = v;
  xcb[idx] = (__bf16)v;
}

// ---------------------------------------------------------------------------
// Selective scan: thread = (batch, channel); 16 states in registers.
// B/C (32 floats per (b,s)) staged to LDS, shared by all channels.
// ---------------------------------------------------------------------------
__global__ __launch_bounds__(256) void scan_kernel(
    const float* __restrict__ u, const float* __restrict__ dt,
    const float* __restrict__ dbl, const float* __restrict__ A_log,
    const float* __restrict__ Dp, float* __restrict__ ys)
{
  int b = blockIdx.x >> 2;
  int d = (blockIdx.x & 3) * 256 + threadIdx.x;
  float a[DST_];
#pragma unroll
  for (int n = 0; n < DST_; ++n) a[n] = -__expf(A_log[(size_t)d * DST_ + n]);
  float Dv = Dp[d];
  float hs[DST_];
#pragma unroll
  for (int n = 0; n < DST_; ++n) hs[n] = 0.f;

  __shared__ float sBC[2 * DST_];
  size_t base = (size_t)b * SEQ_;
  for (int s = 0; s < SEQ_; ++s) {
    size_t row = base + s;
    if (threadIdx.x < 2 * DST_)
      sBC[threadIdx.x] = dbl[row * NPROJ_ + DTR_ + threadIdx.x];
    __syncthreads();
    float dtv = dt[row * DI_ + d];
    float uv  = u[row * DI_ + d];
    float y   = uv * Dv;
#pragma unroll
    for (int n = 0; n < DST_; ++n) {
      float dA = __expf(dtv * a[n]);
      hs[n] = dA * hs[n] + dtv * sBC[n] * uv;
      y += hs[n] * sBC[DST_ + n];
    }
    ys[row * DI_ + d] = y;
    __syncthreads();
  }
}

// ---------------------------------------------------------------------------
// ysb = bf16( ys * silu(z) ),  z = second half of xz rows
// ---------------------------------------------------------------------------
__global__ __launch_bounds__(256) void gate_kernel(
    const float* __restrict__ ys, const float* __restrict__ xz,
    __bf16* __restrict__ ysb)
{
  int idx = blockIdx.x * 256 + threadIdx.x;   // over R_TOT * DI_
  int d   = idx & (DI_ - 1);
  int row = idx >> 10;
  float z = xz[(size_t)row * (2 * DI_) + DI_ + d];
  ysb[idx] = (__bf16)(ys[idx] * silu_f(z));
}

// ---------------------------------------------------------------------------
// Softmax over contiguous groups of 32 (N_CAT), one wave per group, in place
// ---------------------------------------------------------------------------
__global__ __launch_bounds__(256) void softmax32_kernel(float* __restrict__ o)
{
  int g    = blockIdx.x * 8 + (threadIdx.x >> 5);
  int lane = threadIdx.x & 31;
  float* p = o + (size_t)g * 32;
  float v = p[lane];
  float m = v;
#pragma unroll
  for (int off = 16; off > 0; off >>= 1) m = fmaxf(m, __shfl_xor(m, off, 32));
  float e = __expf(v - m);
  float sum = e;
#pragma unroll
  for (int off = 16; off > 0; off >>= 1) sum += __shfl_xor(sum, off, 32);
  p[lane] = e / sum;
}

// ---------------------------------------------------------------------------
extern "C" void kernel_launch(void* const* d_in, const int* in_sizes, int n_in,
                              void* d_out, int out_size, void* d_ws, size_t ws_size,
                              hipStream_t stream)
{
  const float* x      = (const float*)d_in[0];
  const float* lin1_w = (const float*)d_in[1];
  const float* lin1_b = (const float*)d_in[2];
  const float* norm_w = (const float*)d_in[3];
  const float* in_w   = (const float*)d_in[4];
  const float* conv_w = (const float*)d_in[5];
  const float* conv_b = (const float*)d_in[6];
  const float* xprj_w = (const float*)d_in[7];
  const float* dt_w   = (const float*)d_in[8];
  const float* dt_b   = (const float*)d_in[9];
  const float* A_log  = (const float*)d_in[10];
  const float* Dp     = (const float*)d_in[11];
  const float* out_w  = (const float*)d_in[12];
  const float* lin2_w = (const float*)d_in[13];
  const float* lin2_b = (const float*)d_in[14];
  float* out = (float*)d_out;
  (void)ws_size; (void)in_sizes; (void)n_in; (void)out_size;

  // ---- workspace carve (all sizes 16B-aligned) ----
  char* w8 = (char*)d_ws;
  float* h    = (float*)w8; w8 += (size_t)R_TOT * DM_ * 4;       // 8 MB
  float* xz   = (float*)w8; w8 += (size_t)R_TOT * 2 * DI_ * 4;   // 32 MB
  float* xc   = (float*)w8; w8 += (size_t)R_TOT * DI_ * 4;       // 16 MB
  float* dbl  = (float*)w8; w8 += (size_t)R_TOT * NPROJ_ * 4;    // 1 MB
  float* dtb  = (float*)w8; w8 += (size_t)R_TOT * DI_ * 4;       // 16 MB
  float* ysc  = (float*)w8; w8 += (size_t)R_TOT * DI_ * 4;       // 16 MB
  __bf16* xb   = (__bf16*)w8; w8 += (size_t)R_TOT * LAT_ * 2;    // 8 MB
  __bf16* hnb  = (__bf16*)w8; w8 += (size_t)R_TOT * DM_ * 2;     // 4 MB
  __bf16* xcb  = (__bf16*)w8; w8 += (size_t)R_TOT * DI_ * 2;     // 8 MB
  __bf16* dblb = (__bf16*)w8; w8 += (size_t)R_TOT * NPROJ_ * 2;  // 0.5 MB
  __bf16* ysb  = (__bf16*)w8; w8 += (size_t)R_TOT * DI_ * 2;     // 8 MB
  __bf16* hb   = (__bf16*)w8; w8 += (size_t)R_TOT * DM_ * 2;     // 4 MB
  __bf16* wlin1 = (__bf16*)w8; w8 += (size_t)LAT_ * DM_ * 2;
  __bf16* win   = (__bf16*)w8; w8 += (size_t)4 * DM_ * 2 * DI_ * 2;
  __bf16* wxp   = (__bf16*)w8; w8 += (size_t)4 * DI_ * NPROJ_ * 2;
  __bf16* wdt   = (__bf16*)w8; w8 += (size_t)4 * DTR_ * DI_ * 2;
  __bf16* wout  = (__bf16*)w8; w8 += (size_t)4 * DI_ * DM_ * 2;
  __bf16* wlin2 = (__bf16*)w8; w8 += (size_t)DM_ * LAT_ * 2;

  dim3 blk(256);
  const int MB = R_TOT / 128;   // 32 M-tiles

  // ---- one-time bf16 conversions (deterministic, every call) ----
  f2b_kernel<<<(R_TOT * LAT_) / 1024, blk, 0, stream>>>(x, xb);
  f2b_kernel<<<(LAT_ * DM_) / 1024, blk, 0, stream>>>(lin1_w, wlin1);
  f2b_kernel<<<(4 * DM_ * 2 * DI_) / 1024, blk, 0, stream>>>(in_w, win);
  f2b_kernel<<<(4 * DI_ * NPROJ_) / 1024, blk, 0, stream>>>(xprj_w, wxp);
  f2b_kernel<<<(4 * DTR_ * DI_) / 1024, blk, 0, stream>>>(dt_w, wdt);
  f2b_kernel<<<(4 * DI_ * DM_) / 1024, blk, 0, stream>>>(out_w, wout);
  f2b_kernel<<<(DM_ * LAT_) / 1024, blk, 0, stream>>>(lin2_w, wlin2);

  // h = x @ lin1_w + lin1_b   (M=4096, N=512, K=1024)
  gemm_bf16<4, DM_, LAT_, LAT_, 0, false, true>
      <<<dim3(DM_ / 128, MB), blk, 0, stream>>>(xb, wlin1, lin1_b, h);

  for (int l = 0; l < 4; ++l) {
    rmsnorm_kernel<<<R_TOT / 8, blk, 0, stream>>>(h, norm_w + (size_t)l * DM_, hnb);

    // xz = hn @ in_w[l]   (N=2048, K=512)
    gemm_bf16<4, 2 * DI_, DM_, DM_, 0, false, false>
        <<<dim3((2 * DI_) / 128, MB), blk, 0, stream>>>(
            hnb, win + (size_t)l * DM_ * 2 * DI_, nullptr, xz);

    conv_silu_kernel<<<(R_TOT * DI_) / 256, blk, 0, stream>>>(
        xz, conv_w + (size_t)l * DI_ * 4, conv_b + (size_t)l * DI_, xc, xcb);

    // dbl = xc @ xproj_w[l]   (N=64, K=1024)  — BN=64 variant
    gemm_bf16<2, NPROJ_, DI_, DI_, 0, false, false>
        <<<dim3(1, MB), blk, 0, stream>>>(
            xcb, wxp + (size_t)l * DI_ * NPROJ_, nullptr, dbl);

    f2b_kernel<<<(R_TOT * NPROJ_) / 1024, blk, 0, stream>>>(dbl, dblb);

    // dt = softplus(dbl[:, :32] @ dt_w[l] + dt_b[l])   (N=1024, K=32, lda=64)
    gemm_bf16<4, DI_, DTR_, NPROJ_, 1, false, true>
        <<<dim3(DI_ / 128, MB), blk, 0, stream>>>(
            dblb, wdt + (size_t)l * DTR_ * DI_, dt_b + (size_t)l * DI_, dtb);

    scan_kernel<<<BATCH_ * (DI_ / 256), blk, 0, stream>>>(
        xc, dtb, dbl, A_log + (size_t)l * DI_ * DST_, Dp + (size_t)l * DI_, ysc);

    gate_kernel<<<(R_TOT * DI_) / 256, blk, 0, stream>>>(ysc, xz, ysb);

    // h += ysb @ out_w[l]   (N=512, K=1024, accumulate)
    gemm_bf16<4, DM_, DI_, DI_, 0, true, false>
        <<<dim3(DM_ / 128, MB), blk, 0, stream>>>(
            ysb, wout + (size_t)l * DI_ * DM_, nullptr, h);
  }

  // out = h @ lin2_w + lin2_b   (N=1024, K=512), then group softmax in place
  f2b_kernel<<<(R_TOT * DM_) / 1024, blk, 0, stream>>>(h, hb);
  gemm_bf16<4, LAT_, DM_, DM_, 0, false, true>
      <<<dim3(LAT_ / 128, MB), blk, 0, stream>>>(hb, wlin2, lin2_b, out);
  softmax32_kernel<<<(R_TOT * 32) / 8, blk, 0, stream>>>(out);
}
